// GraphAttentionEmbedding_11416023072997
// MI455X (gfx1250) — matrix operand order repaired and checked by reading, not compiled, not run on hardware
//
#include <hip/hip_runtime.h>
#include <hip/hip_fp16.h>

// MI455X / gfx1250, wave32. D=128, H=2, DH=64.

typedef _Float16 h16v __attribute__((ext_vector_type(16)));
typedef _Float16 h8v  __attribute__((ext_vector_type(8)));
typedef float    f8v  __attribute__((ext_vector_type(8)));

union AFrag { h16v v; h8v h[2]; };

__device__ inline f8v splat8(float b) {
    f8v r;
#pragma unroll
    for (int i = 0; i < 8; ++i) r[i] = b;
    return r;
}

// A fragment (16x32 f16, M x K) from an LDS tile stored row-major [16][128] f16.
// Lane layout: M = lane&15; lane<16 holds K = kb+{0..7, 16..23}, lane>=16 holds K = kb+{8..15, 24..31}.
__device__ inline h16v ldsA(const _Float16* tile, int lane, int kb) {
    const int m  = lane & 15;
    const int hi = lane >> 4;
    AFrag a;
    a.h[0] = *(const h8v*)(tile + m * 128 + kb + hi * 8);
    a.h[1] = *(const h8v*)(tile + m * 128 + kb + 16 + hi * 8);
    return a.v;
}

// B fragment (32x16 f16, K x N) from a column-major f16 weight copy Wc[n*stride + k].
// Lane layout: N = lane&15 fixed per lane; lane<16 holds K = kb+{0..7, 16..23}, lane>=16 holds K = kb+{8..15, 24..31}.
__device__ inline h16v gblB(const _Float16* Wc, int stride, int ncol, int kb, int lane) {
    const int hi = lane >> 4;
    const _Float16* p = Wc + (size_t)ncol * stride + kb + hi * 8;
    AFrag b;
    b.h[0] = *(const h8v*)(p);
    b.h[1] = *(const h8v*)(p + 16);
    return b.v;
}

__device__ inline f8v wmma_f16(h16v a, h16v b, f8v c) {
    // D = A*B + C, f32 accumulate (v_wmma_f32_16x16x32_f16)
    return __builtin_amdgcn_wmma_f32_16x16x32_f16(false, a, false, b, (short)0, c, false, false);
}

__device__ inline void atomicAddF32(float* p, float v) {
    __hip_atomic_fetch_add(p, v, __ATOMIC_RELAXED, __HIP_MEMORY_SCOPE_AGENT);
}

// ---------------------------------------------------------------------------
// Kernel A: convert weights to f16 column-major copies for direct B-fragment
// loads. WC = [Wq|Wk|Wv|Wskip] each 128x128 -> Wc[n*128+k]; WeC 256x128 -> [n*256+k].
// ---------------------------------------------------------------------------
__global__ __launch_bounds__(256) void convert_weights_kernel(
    const float* __restrict__ Wq, const float* __restrict__ Wk,
    const float* __restrict__ Wv, const float* __restrict__ Ws,
    const float* __restrict__ We,
    _Float16* __restrict__ WC, _Float16* __restrict__ WeC)
{
    const int g = blockIdx.x * 256 + threadIdx.x;
    if (g < 4 * 16384) {
        const int m = g >> 14, idx = g & 16383;
        const int kk = idx >> 7, nn = idx & 127;
        const float* W = (m == 0) ? Wq : (m == 1) ? Wk : (m == 2) ? Wv : Ws;
        WC[m * 16384 + nn * 128 + kk] = (_Float16)W[kk * 128 + nn];
    } else {
        const int g2 = g - 4 * 16384;           // < 32768
        const int kk = g2 >> 7, nn = g2 & 127;  // kk in [0,256)
        WeC[nn * 256 + kk] = (_Float16)We[kk * 128 + nn];
    }
}

// ---------------------------------------------------------------------------
// Kernel B: node projections. One block = 16 node rows; 8 waves cover the 128
// output columns; each wave runs all 4 weight matrices sharing the A fragment.
// ---------------------------------------------------------------------------
__global__ __launch_bounds__(256) void node_proj_kernel(
    const float* __restrict__ x, const _Float16* __restrict__ WC,
    const float* __restrict__ bq, const float* __restrict__ bk,
    const float* __restrict__ bv, const float* __restrict__ bskip,
    float* __restrict__ qw, float* __restrict__ kw, float* __restrict__ vw,
    float* __restrict__ outw)
{
    __shared__ __align__(16) _Float16 Xt[16 * 128];
    const int tid = threadIdx.x;
    const int rowbase = blockIdx.x * 16;

    {   // stage 16x128 x-tile to LDS as f16 (coalesced, 8 floats/thread)
        const int i = tid >> 4;
        const int d = (tid & 15) * 8;
        const float* src = x + (size_t)(rowbase + i) * 128 + d;
        _Float16* dst = Xt + i * 128 + d;
#pragma unroll
        for (int j = 0; j < 8; ++j) dst[j] = (_Float16)src[j];
    }
    __syncthreads();

    const int lane = tid & 31;
    const int wave = tid >> 5;
    const int n0   = wave * 16;
    const int ncol = n0 + (lane & 15);

    const float* biases[4] = { bq, bk, bv, bskip };
    float* outs[4]         = { qw, kw, vw, outw };

    f8v acc[4];
#pragma unroll
    for (int w = 0; w < 4; ++w) acc[w] = splat8(biases[w][ncol]);

#pragma unroll
    for (int ks = 0; ks < 4; ++ks) {
        const h16v a = ldsA(Xt, lane, ks * 32);
#pragma unroll
        for (int w = 0; w < 4; ++w) {
            const h16v b = gblB(WC + (size_t)w * 16384, 128, ncol, ks * 32, lane);
            acc[w] = wmma_f16(a, b, acc[w]);
        }
    }

    const int hi = lane >> 4;
#pragma unroll
    for (int w = 0; w < 4; ++w) {
        float* o = outs[w];
#pragma unroll
        for (int r = 0; r < 8; ++r) {
            const int m = r + hi * 8;           // C layout: vgpr r -> M=r (lanes<16), M=8+r (lanes>=16)
            o[(size_t)(rowbase + m) * 128 + ncol] = acc[w][r];
        }
    }
}

// ---------------------------------------------------------------------------
// Kernel C: fused edge pass. One block = 16 edges.
//  time_enc = cos(rel_t*Wt+bt); e = time_enc@We_top + msg@We_bot (WMMA);
//  alpha = q_dst . (k_src + e) / 8; ex = exp(alpha);
//  atomically accumulate ex and ex*(v_src + e) per destination node.
// ---------------------------------------------------------------------------
__global__ __launch_bounds__(256) void edge_attn_kernel(
    const int* __restrict__ ei, int E,
    const float* __restrict__ last_update, const float* __restrict__ tarr,
    const float* __restrict__ msg,
    const float* __restrict__ Wt, const float* __restrict__ bt,
    const _Float16* __restrict__ WeC,
    const float* __restrict__ qw, const float* __restrict__ kw,
    const float* __restrict__ vw,
    float* __restrict__ numer, float* __restrict__ denom)
{
    __shared__ __align__(16) _Float16 Tt[16 * 128];
    __shared__ __align__(16) _Float16 Mt[16 * 128];
    __shared__ __align__(16) float    Et[16 * 128];
    __shared__ int   srcs[16];
    __shared__ int   dsts[16];
    __shared__ float relts[16];

    const int tid = threadIdx.x;
    const int eb  = blockIdx.x * 16;

    if (tid < 16) {
        const int s = ei[eb + tid];
        const int d = ei[E + eb + tid];
        srcs[tid]  = s;
        dsts[tid]  = d;
        relts[tid] = last_update[s] - tarr[eb + tid];
    }
    // prefetch next tile's msg rows into cache (global_prefetch_b8)
    if (eb + 16 < E) __builtin_prefetch(msg + (size_t)(eb + 16) * 128 + tid * 8, 0, 1);

    {   // stage msg tile as f16
        const int i = tid >> 4;
        const int d = (tid & 15) * 8;
        const float* src = msg + (size_t)(eb + i) * 128 + d;
        _Float16* dst = Mt + i * 128 + d;
#pragma unroll
        for (int j = 0; j < 8; ++j) dst[j] = (_Float16)src[j];
    }
    __syncthreads();

    {   // time encoding tile as f16
        const int i = tid >> 4;
        const int d = (tid & 15) * 8;
        const float rt = relts[i];
        _Float16* dst = Tt + i * 128 + d;
#pragma unroll
        for (int j = 0; j < 8; ++j)
            dst[j] = (_Float16)__cosf(rt * Wt[d + j] + bt[d + j]);
    }
    __syncthreads();

    const int lane = tid & 31;
    const int wave = tid >> 5;
    const int n0   = wave * 16;
    const int ncol = n0 + (lane & 15);

    f8v acc = splat8(0.0f);
#pragma unroll
    for (int ks = 0; ks < 4; ++ks) {        // time-enc part: We rows 0..127
        const h16v a = ldsA(Tt, lane, ks * 32);
        const h16v b = gblB(WeC, 256, ncol, ks * 32, lane);
        acc = wmma_f16(a, b, acc);
    }
#pragma unroll
    for (int ks = 0; ks < 4; ++ks) {        // msg part: We rows 128..255
        const h16v a = ldsA(Mt, lane, ks * 32);
        const h16v b = gblB(WeC, 256, ncol, 128 + ks * 32, lane);
        acc = wmma_f16(a, b, acc);
    }

    const int hi = lane >> 4;
#pragma unroll
    for (int r = 0; r < 8; ++r) Et[(r + hi * 8) * 128 + ncol] = acc[r];
    __syncthreads();

    // --- attention: 16 threads per edge, 8 dims each; heads split at d=64 ---
    const int i  = tid >> 4;
    const int s  = tid & 15;
    const int d0 = s * 8;
    const int srcn = srcs[i], dstn = dsts[i];
    const float* qrow = qw + (size_t)dstn * 128;
    const float* krow = kw + (size_t)srcn * 128;
    const float* vrow = vw + (size_t)srcn * 128;
    const float* erow = Et + i * 128;

    float p = 0.0f;
#pragma unroll
    for (int j = 0; j < 8; ++j) {
        const int d = d0 + j;
        p += qrow[d] * (krow[d] + erow[d]);
    }
    // butterfly reduce within the 8-lane head group (all lanes get the sum)
    p += __shfl_xor(p, 1, 8);
    p += __shfl_xor(p, 2, 8);
    p += __shfl_xor(p, 4, 8);

    const float ex = __expf(p * 0.125f);    // 1/sqrt(DH)=1/8; alpha is O(1), max-shift unneeded
    const int head = s >> 3;
    if ((s & 7) == 0) atomicAddF32(&denom[(size_t)dstn * 2 + head], ex);

    float* nrow = numer + (size_t)dstn * 128;
#pragma unroll
    for (int j = 0; j < 8; ++j) {
        const int d = d0 + j;
        atomicAddF32(&nrow[d], ex * (vrow[d] + erow[d]));
    }
}

// ---------------------------------------------------------------------------
// Kernel D: out = skip (already in out) + numer/denom  (guard empty segments)
// ---------------------------------------------------------------------------
__global__ __launch_bounds__(256) void finalize_kernel(
    float* __restrict__ out, const float* __restrict__ numer,
    const float* __restrict__ denom)
{
    const size_t g = (size_t)blockIdx.x * 256 + threadIdx.x;
    const int n = (int)(g >> 7);
    const int d = (int)(g & 127);
    const int h = d >> 6;
    const float dn = denom[(size_t)n * 2 + h];
    const float c = (dn > 0.0f) ? numer[g] / dn : 0.0f;
    out[g] += c;
}

extern "C" void kernel_launch(void* const* d_in, const int* in_sizes, int n_in,
                              void* d_out, int out_size, void* d_ws, size_t ws_size,
                              hipStream_t stream)
{
    const float* x   = (const float*)d_in[0];
    const float* lu  = (const float*)d_in[1];
    const int*   ei  = (const int*)  d_in[2];
    const float* t   = (const float*)d_in[3];
    const float* msg = (const float*)d_in[4];
    const float* Wt  = (const float*)d_in[5];
    const float* bt  = (const float*)d_in[6];
    const float* Wq  = (const float*)d_in[7];
    const float* bq  = (const float*)d_in[8];
    const float* Wk  = (const float*)d_in[9];
    const float* bk  = (const float*)d_in[10];
    const float* Wv  = (const float*)d_in[11];
    const float* bv  = (const float*)d_in[12];
    const float* We  = (const float*)d_in[13];
    const float* Ws  = (const float*)d_in[14];
    const float* bs  = (const float*)d_in[15];
    float* out = (float*)d_out;

    const int N = in_sizes[0] / 128;   // 50000
    const int E = in_sizes[2] / 2;     // 800000

    char* ws = (char*)d_ws;
    size_t off = 0;
    auto alloc = [&](size_t bytes) {
        size_t o = off;
        off = (off + bytes + 255) & ~(size_t)255;
        return o;
    };
    const size_t qOff   = alloc((size_t)N * 128 * 4);
    const size_t kOff   = alloc((size_t)N * 128 * 4);
    const size_t vOff   = alloc((size_t)N * 128 * 4);
    const size_t numOff = alloc((size_t)N * 128 * 4);   // numer; denom follows contiguously
    const size_t denOff = alloc((size_t)N * 2 * 4);
    const size_t wcOff  = alloc((size_t)4 * 16384 * 2); // f16 Wq|Wk|Wv|Wskip (col-major)
    const size_t wecOff = alloc((size_t)32768 * 2);     // f16 We (col-major, stride 256)
    (void)ws_size; (void)n_in; (void)out_size;

    float* qw = (float*)(ws + qOff);
    float* kw = (float*)(ws + kOff);
    float* vw = (float*)(ws + vOff);
    float* numer = (float*)(ws + numOff);
    float* denom = (float*)(ws + denOff);
    _Float16* WC  = (_Float16*)(ws + wcOff);
    _Float16* WeC = (_Float16*)(ws + wecOff);

    // zero the accumulators (numer+denom are contiguous)
    hipMemsetAsync(ws + numOff, 0, (size_t)N * 128 * 4 + (size_t)N * 2 * 4, stream);

    convert_weights_kernel<<<(4 * 16384 + 32768) / 256, 256, 0, stream>>>(
        Wq, Wk, Wv, Ws, We, WC, WeC);

    node_proj_kernel<<<N / 16, 256, 0, stream>>>(
        x, WC, bq, bk, bv, bs, qw, kw, vw, out);

    edge_attn_kernel<<<E / 16, 256, 0, stream>>>(
        ei, E, lu, t, msg, Wt, bt, WeC, qw, kw, vw, numer, denom);

    finalize_kernel<<<(N * 128) / 256, 256, 0, stream>>>(out, numer, denom);
}
